// CharDecoder_86706799772576
// MI455X (gfx1250) — matrix operand order repaired and checked by reading, not compiled, not run on hardware
//
#include <hip/hip_runtime.h>

// ---------------- problem sizes ----------------
#define LSEQ 64
#define BATCH 2048
#define HID 1024
#define EMB 50
#define EPAD 64              // E padded to WMMA K granularity
#define VOC 256
#define NKT 34               // (EPAD + HID) / 32 K-tiles for gate GEMM
#define NKT_P 32             // HID / 32 K-tiles for projection GEMM
#define LDSP 40              // padded LDS row stride (u16) to avoid bank conflicts

typedef unsigned short u16_t;
typedef unsigned int   u32_t;
typedef __attribute__((ext_vector_type(8)))  unsigned short us8;
typedef __attribute__((ext_vector_type(16))) unsigned short us16;
typedef __attribute__((ext_vector_type(16))) __bf16        v16bf;
typedef __attribute__((ext_vector_type(8)))  float         v8f;

union Frag16 {
  u16_t h[16];
  us8   half2[2];
  us16  full;
  v16bf v;
};

__device__ __forceinline__ u16_t f2bf(float f) {
  u32_t x = __float_as_uint(f);
  u32_t r = x + 0x7FFFu + ((x >> 16) & 1u);   // round-to-nearest-even
  return (u16_t)(r >> 16);
}
__device__ __forceinline__ float sigf(float x) {
  return 1.0f / (1.0f + __expf(-x));
}

// ---- async global->LDS staging (CDNA5 ASYNCcnt path, ISA 08 §4 / §15.18.3) ----
// Copies 32 bytes per lane: mem[src .. +31] -> LDS[dst .. +31], no VGPR data.
// VDST = 32-bit LDS byte offset (= low 32 bits of generic LDS pointer per the
// aperture mapping); inst offset applies to BOTH global and LDS addresses.
__device__ __forceinline__ void stage_async32(const u16_t* src, u16_t* ldsdst) {
  const u32_t l = (u32_t)(uintptr_t)ldsdst;
  asm volatile("global_load_async_to_lds_b128 %0, %1, off"
               :: "v"(l), "v"(src) : "memory");
  asm volatile("global_load_async_to_lds_b128 %0, %1, off offset:16"
               :: "v"(l), "v"(src) : "memory");
}
__device__ __forceinline__ void wait_async0() {
  asm volatile("s_wait_asynccnt 0x0" ::: "memory");
}

// ---------------- prep: pack [W_ih|W_hh] into B-fragment layout ----------------
// fragment (nt, kt): lane L holds N = nt*16 + (L&15), K = kt*32 + (L>>4)*8 + (j&7) + (j>>3)*16
__global__ void pack_gate_weights(const float* __restrict__ Wih,
                                  const float* __restrict__ Whh,
                                  u16_t* __restrict__ Wp) {
  const int nt   = blockIdx.x;   // 0..255  (N = 4096)
  const int kt   = blockIdx.y;   // 0..33   (K = 1088)
  const int lane = threadIdx.x;  // 0..31
  const int n  = nt * 16 + (lane & 15);
  const int kh = (lane >> 4) * 8;
  u16_t* dst = Wp + (((size_t)nt * NKT + kt) * 32 + lane) * 16;
#pragma unroll
  for (int j = 0; j < 16; ++j) {
    const int k = kt * 32 + kh + (j & 7) + (j >> 3) * 16;
    float v;
    if (k < EPAD) v = (k < EMB) ? Wih[(size_t)n * EMB + k] : 0.0f;
    else          v = Whh[(size_t)n * HID + (k - EPAD)];
    dst[j] = f2bf(v);
  }
}

__global__ void pack_proj_weights(const float* __restrict__ Wprj,
                                  u16_t* __restrict__ Pp) {
  const int nt   = blockIdx.x;   // 0..15  (N = 256)
  const int kt   = blockIdx.y;   // 0..31  (K = 1024)
  const int lane = threadIdx.x;
  const int n  = nt * 16 + (lane & 15);
  const int kh = (lane >> 4) * 8;
  u16_t* dst = Pp + (((size_t)nt * NKT_P + kt) * 32 + lane) * 16;
#pragma unroll
  for (int j = 0; j < 16; ++j)
    dst[j] = f2bf(Wprj[(size_t)n * HID + kt * 32 + kh + (j & 7) + (j >> 3) * 16]);
}

// ---------------- prep: embedding gather + cast, pad E -> 64 ----------------
__global__ void embed_cast(const int* __restrict__ tok,
                           const float* __restrict__ emb,
                           u16_t* __restrict__ x) {
  const size_t g = (size_t)blockIdx.x * blockDim.x + threadIdx.x; // < L*B*EPAD
  const int   k  = (int)(g & (EPAD - 1));
  const size_t lb = g >> 6;
  const int   t  = tok[lb];
  const float v  = (k < EMB) ? emb[(size_t)t * EMB + k] : 0.0f;
  x[g] = f2bf(v);
}

// ---------------- prep: h0 -> bf16, c0 -> fp32 ws, fused bias ----------------
__global__ void init_state(const float* __restrict__ h0,
                           const float* __restrict__ c0,
                           const float* __restrict__ bih,
                           const float* __restrict__ bhh,
                           u16_t* __restrict__ hA,
                           float* __restrict__ c,
                           float* __restrict__ bias) {
  const size_t g = (size_t)blockIdx.x * blockDim.x + threadIdx.x; // < B*H
  hA[g] = f2bf(h0[g]);
  c[g]  = c0[g];
  if (g < 4 * HID) bias[g] = bih[g] + bhh[g];
}

// ---------------- LSTM step: gates GEMM + fused pointwise ----------------
// grid (B/128, H/32), block 256 = 8 waves (4 in M x 2 in N).
// Block tile: 128 rows x 32 cols per gate; each wave holds two 16-row sub-tiles
// x 16 cols x 4 gates = 8 f32 accumulators; every B fragment feeds 2 WMMAs.
// A-tile double-buffered in LDS, staged by async global->LDS DMA one iteration
// ahead (ASYNCcnt), single barrier per K-iteration.
__global__ __launch_bounds__(256)
void lstm_step(const u16_t* __restrict__ xt,    // (B, EPAD) bf16, step t
               const u16_t* __restrict__ hin,   // (B, H) bf16
               u16_t* __restrict__ hout,        // (B, H) bf16
               float* __restrict__ c,           // (B, H) f32, in-place
               const u16_t* __restrict__ Wp,    // packed gate weights
               const float* __restrict__ bias,  // (4H) f32
               float* __restrict__ hT,
               float* __restrict__ cT,
               int last) {
  __shared__ u16_t sA[2][128 * LDSP];           // 2 x 10 KB double buffer

  const int tid  = threadIdx.x;
  const int lane = tid & 31;
  const int wave = tid >> 5;
  const int wm   = wave >> 1;                   // 0..3 (M quarter, 32 rows)
  const int wn   = wave & 1;                    // 0..1 (N sub-tile)
  const int b0   = blockIdx.x * 128;
  const int nb0  = blockIdx.y * 32;             // per-gate column base

  v8f a0i = {}, a0f = {}, a0g = {}, a0o = {};   // sub-tile 0 accumulators
  v8f a1i = {}, a1f = {}, a1g = {}, a1o = {};   // sub-tile 1 accumulators

  // cooperative A stager: 256 threads x 32B cover 128x32 bf16 per K-iter
  const int lrow = tid >> 1;                    // 0..127
  const int lseg = tid & 1;                     // 0..1 (16 bf16 each)
  const int soff = lrow * LDSP + lseg * 16;

  // A-fragment read offsets (16-bit A layout: lane -> M, K-half split)
  const int arow0 = wm * 32 + (lane & 15);      // sub-tile 0 row
  const int akh   = (lane >> 4) * 8;
  const int aoff0 = arow0 * LDSP + akh;         // + (0|16) for the two K chunks
  const int aoff1 = aoff0 + 16 * LDSP;          // sub-tile 1 row (+16 rows)

  // per-gate packed-B pointers (advance 512 u16 per K-tile)
  const int ntbase = blockIdx.y * 2 + wn;
  const u16_t* bp0 = Wp + (size_t)(0 * 64 + ntbase) * NKT * 512 + (size_t)lane * 16;
  const u16_t* bp1 = Wp + (size_t)(1 * 64 + ntbase) * NKT * 512 + (size_t)lane * 16;
  const u16_t* bp2 = Wp + (size_t)(2 * 64 + ntbase) * NKT * 512 + (size_t)lane * 16;
  const u16_t* bp3 = Wp + (size_t)(3 * 64 + ntbase) * NKT * 512 + (size_t)lane * 16;

  // prologue: stage K-tile 0 (x-part)
  stage_async32(xt + (size_t)(b0 + lrow) * EPAD + lseg * 16, &sA[0][soff]);

  for (int kt = 0; kt < NKT; ++kt) {
    wait_async0();                              // my tile-kt DMA complete
    __syncthreads();                            // all tiles landed; old reads done

    if (kt + 1 < NKT) {                         // stage next tile into other buf
      const int kn = kt + 1;
      const u16_t* nsrc = (kn < 2)
          ? xt  + (size_t)(b0 + lrow) * EPAD + kn * 32 + lseg * 16
          : hin + (size_t)(b0 + lrow) * HID + (kn * 32 - EPAD) + lseg * 16;
      stage_async32(nsrc, &sA[kn & 1][soff]);
    }

    const u16_t* buf = sA[kt & 1];
    Frag16 fa0, fa1;
    fa0.half2[0] = *(const us8*)(buf + aoff0);
    fa0.half2[1] = *(const us8*)(buf + aoff0 + 16);
    fa1.half2[0] = *(const us8*)(buf + aoff1);
    fa1.half2[1] = *(const us8*)(buf + aoff1 + 16);

    Frag16 b;
    b.full = *(const us16*)bp0;  bp0 += 512;
    a0i = __builtin_amdgcn_wmma_f32_16x16x32_bf16(false, fa0.v, false, b.v, (short)0, a0i, false, false);
    a1i = __builtin_amdgcn_wmma_f32_16x16x32_bf16(false, fa1.v, false, b.v, (short)0, a1i, false, false);
    b.full = *(const us16*)bp1;  bp1 += 512;
    a0f = __builtin_amdgcn_wmma_f32_16x16x32_bf16(false, fa0.v, false, b.v, (short)0, a0f, false, false);
    a1f = __builtin_amdgcn_wmma_f32_16x16x32_bf16(false, fa1.v, false, b.v, (short)0, a1f, false, false);
    b.full = *(const us16*)bp2;  bp2 += 512;
    a0g = __builtin_amdgcn_wmma_f32_16x16x32_bf16(false, fa0.v, false, b.v, (short)0, a0g, false, false);
    a1g = __builtin_amdgcn_wmma_f32_16x16x32_bf16(false, fa1.v, false, b.v, (short)0, a1g, false, false);
    b.full = *(const us16*)bp3;  bp3 += 512;
    a0o = __builtin_amdgcn_wmma_f32_16x16x32_bf16(false, fa0.v, false, b.v, (short)0, a0o, false, false);
    a1o = __builtin_amdgcn_wmma_f32_16x16x32_bf16(false, fa1.v, false, b.v, (short)0, a1o, false, false);

    __builtin_prefetch(bp0, 0, 1);              // global_prefetch next weight tile
  }

  // fused LSTM pointwise on D fragments (C/D layout: lane->N, vgpr->M)
  const int n = nb0 + wn * 16 + (lane & 15);
  const float bi = bias[n];
  const float bf = bias[HID + n];
  const float bg = bias[2 * HID + n];
  const float bo = bias[3 * HID + n];
#pragma unroll
  for (int sub = 0; sub < 2; ++sub) {
    const v8f& ai = sub ? a1i : a0i;
    const v8f& af = sub ? a1f : a0f;
    const v8f& ag = sub ? a1g : a0g;
    const v8f& ao = sub ? a1o : a0o;
    const int mb = b0 + wm * 32 + sub * 16 + (lane >> 4) * 8;
#pragma unroll
    for (int r = 0; r < 8; ++r) {
      const size_t idx = (size_t)(mb + r) * HID + n;
      const float iv = sigf(ai[r] + bi);
      const float fv = sigf(af[r] + bf);
      const float gv = tanhf(ag[r] + bg);
      const float ov = sigf(ao[r] + bo);
      const float cn = fv * c[idx] + iv * gv;
      const float hn = ov * tanhf(cn);
      c[idx]    = cn;
      hout[idx] = f2bf(hn);
      if (last) { hT[idx] = hn; cT[idx] = cn; }
    }
  }
}

// ---------------- per-step projection: scores[t] = h @ W_proj.T + b ----------------
// grid (B/128, V/32), same 128-row / async double-buffer structure.
__global__ __launch_bounds__(256)
void proj_step(const u16_t* __restrict__ h,     // (B, H) bf16
               const u16_t* __restrict__ Pp,    // packed proj weights
               const float* __restrict__ bprj,  // (V)
               float* __restrict__ out) {       // (B, V) slab for step t
  __shared__ u16_t sA[2][128 * LDSP];

  const int tid  = threadIdx.x;
  const int lane = tid & 31;
  const int wave = tid >> 5;
  const int wm   = wave >> 1;
  const int wn   = wave & 1;
  const int b0   = blockIdx.x * 128;
  const int n0   = blockIdx.y * 32;

  v8f acc0 = {}, acc1 = {};

  const int lrow = tid >> 1;
  const int lseg = tid & 1;
  const int soff = lrow * LDSP + lseg * 16;

  const int arow0 = wm * 32 + (lane & 15);
  const int akh   = (lane >> 4) * 8;
  const int aoff0 = arow0 * LDSP + akh;
  const int aoff1 = aoff0 + 16 * LDSP;

  const int ntile = blockIdx.y * 2 + wn;
  const u16_t* bp = Pp + (size_t)ntile * NKT_P * 512 + (size_t)lane * 16;

  const u16_t* arow_src = h + (size_t)(b0 + lrow) * HID + lseg * 16;
  stage_async32(arow_src, &sA[0][soff]);        // prologue: K-tile 0

  for (int kt = 0; kt < NKT_P; ++kt) {
    wait_async0();
    __syncthreads();

    if (kt + 1 < NKT_P)
      stage_async32(arow_src + (kt + 1) * 32, &sA[(kt + 1) & 1][soff]);

    const u16_t* buf = sA[kt & 1];
    Frag16 fa0, fa1;
    fa0.half2[0] = *(const us8*)(buf + aoff0);
    fa0.half2[1] = *(const us8*)(buf + aoff0 + 16);
    fa1.half2[0] = *(const us8*)(buf + aoff1);
    fa1.half2[1] = *(const us8*)(buf + aoff1 + 16);

    Frag16 b;
    b.full = *(const us16*)bp;  bp += 512;
    acc0 = __builtin_amdgcn_wmma_f32_16x16x32_bf16(false, fa0.v, false, b.v, (short)0, acc0, false, false);
    acc1 = __builtin_amdgcn_wmma_f32_16x16x32_bf16(false, fa1.v, false, b.v, (short)0, acc1, false, false);
  }

  const int n = n0 + wn * 16 + (lane & 15);
  const float bv = bprj[n];
#pragma unroll
  for (int sub = 0; sub < 2; ++sub) {
    const v8f& acc = sub ? acc1 : acc0;
    const int mb = b0 + wm * 32 + sub * 16 + (lane >> 4) * 8;
#pragma unroll
    for (int r = 0; r < 8; ++r)
      out[(size_t)(mb + r) * VOC + n] = acc[r] + bv;
  }
}

// ---------------- host orchestration ----------------
extern "C" void kernel_launch(void* const* d_in, const int* in_sizes, int n_in,
                              void* d_out, int out_size, void* d_ws, size_t ws_size,
                              hipStream_t stream) {
  (void)in_sizes; (void)n_in; (void)out_size; (void)ws_size;

  const int*   tok  = (const int*)  d_in[0];
  const float* h0   = (const float*)d_in[1];
  const float* c0   = (const float*)d_in[2];
  const float* emb  = (const float*)d_in[3];
  const float* Wih  = (const float*)d_in[4];
  const float* Whh  = (const float*)d_in[5];
  const float* bih  = (const float*)d_in[6];
  const float* bhh  = (const float*)d_in[7];
  const float* Wprj = (const float*)d_in[8];
  const float* bprj = (const float*)d_in[9];

  float* scores = (float*)d_out;                       // (L, B, V)
  float* hT = scores + (size_t)LSEQ * BATCH * VOC;     // (B, H)
  float* cT = hT + (size_t)BATCH * HID;                // (B, H)

  char*  ws  = (char*)d_ws;
  size_t off = 0;
  auto carve = [&](size_t bytes) -> void* {
    void* p = ws + off;
    off += (bytes + 255) & ~(size_t)255;
    return p;
  };
  u16_t* x    = (u16_t*)carve((size_t)LSEQ * BATCH * EPAD * 2);   // 16.8 MB
  u16_t* Wp   = (u16_t*)carve((size_t)256 * NKT * 512 * 2);       //  8.9 MB
  u16_t* Pp   = (u16_t*)carve((size_t)16 * NKT_P * 512 * 2);      //  0.5 MB
  u16_t* hA   = (u16_t*)carve((size_t)BATCH * HID * 2);           //  4.2 MB
  u16_t* hB   = (u16_t*)carve((size_t)BATCH * HID * 2);           //  4.2 MB
  float* cbuf = (float*)carve((size_t)BATCH * HID * 4);           //  8.4 MB
  float* bias = (float*)carve((size_t)4 * HID * 4);

  pack_gate_weights<<<dim3(256, NKT), 32, 0, stream>>>(Wih, Whh, Wp);
  pack_proj_weights<<<dim3(16, NKT_P), 32, 0, stream>>>(Wprj, Pp);
  embed_cast<<<(LSEQ * BATCH * EPAD) / 256, 256, 0, stream>>>(tok, emb, x);
  init_state<<<(BATCH * HID) / 256, 256, 0, stream>>>(h0, c0, bih, bhh, hA, cbuf, bias);

  u16_t* hin = hA;
  u16_t* hout = hB;
  for (int t = 0; t < LSEQ; ++t) {
    lstm_step<<<dim3(BATCH / 128, HID / 32), 256, 0, stream>>>(
        x + (size_t)t * BATCH * EPAD, hin, hout, cbuf, Wp, bias, hT, cT,
        (t == LSEQ - 1) ? 1 : 0);
    proj_step<<<dim3(BATCH / 128, VOC / 32), 256, 0, stream>>>(
        hout, Pp, bprj, scores + (size_t)t * BATCH * VOC);
    u16_t* tmp = hin; hin = hout; hout = tmp;
  }
}
